// RotaryPEMultiHeadAttention_16698832847295
// MI455X (gfx1250) — matrix operand order repaired and checked by reading, not compiled, not run on hardware
//
#include <hip/hip_runtime.h>

// ---------------------------------------------------------------------------
// RoPE multi-head attention, CDNA5 (gfx1250, wave32).
//   f32->f16 pre-convert (X, Wq, Wk, Wv, Wo)
//   -> 3x [A @ W^T] WMMA GEMM (async-LDS double-buffered, RoPE fused for Q,K)
//   -> flash attention (K tile via TDM tensor_load_to_lds, V tile via
//      global_load_async_to_lds_b128, WMMA QK^T and PV, online softmax)
//   -> [O @ Wo^T] WMMA GEMM, f32 output.
// ---------------------------------------------------------------------------

#define HEADS 16
#define DK    64
#define B_    4
#define S_    2048
#define D_    1024
#define M_    (B_ * S_)          // 8192 flattened rows

typedef _Float16 half_t;
typedef __attribute__((ext_vector_type(16))) _Float16 v16h;
typedef __attribute__((ext_vector_type(8)))  _Float16 v8h;
typedef __attribute__((ext_vector_type(4)))  _Float16 v4h;
typedef __attribute__((ext_vector_type(8)))  float    v8f;
typedef __attribute__((ext_vector_type(4)))  unsigned int u32x4;
typedef __attribute__((ext_vector_type(4)))  int          i32x4;
typedef __attribute__((ext_vector_type(8)))  int          i32x8;

__device__ __forceinline__ v8f wmma_f32_f16(v16h a, v16h b, v8f c) {
  return __builtin_amdgcn_wmma_f32_16x16x32_f16(false, a, false, b, (short)0, c,
                                                false, false);
}

// Async global -> LDS, 16 bytes per lane (ASYNCcnt).  VDST = LDS byte address
// (low 32 bits of the generic pointer ARE the LDS offset per the aperture
// mapping), VADDR = 64-bit global address.
__device__ __forceinline__ void async_cp16(const half_t* g, half_t* l) {
  asm volatile("global_load_async_to_lds_b128 %0, %1, off"
               :: "v"((unsigned)(unsigned long long)l),
                  "v"((unsigned long long)g)
               : "memory");
}
__device__ __forceinline__ void wait_async0() {
  asm volatile("s_wait_asynccnt 0x0" ::: "memory");
}

// TDM: load a rows x cols tile of f16 (row pitch srcPitch elements) from
// global to contiguous LDS.  2-D descriptor, data_size=2B, no padding.
// 6-arg builtin (clang-23 / therock-10.0): (g0, g1, g2, g3, g4, cpol).
__device__ __forceinline__ void tdm_load_tile_f16(half_t* lds, const half_t* gsrc,
                                                  unsigned rows, unsigned cols,
                                                  unsigned srcPitch) {
  const unsigned long long ga = (unsigned long long)gsrc;
  u32x4 g0;
  g0[0] = 1u;                                       // count=1 (valid), user mode
  g0[1] = (unsigned)(unsigned long long)lds;        // lds_addr
  g0[2] = (unsigned)ga;                             // global_addr[31:0]
  g0[3] = (unsigned)((ga >> 32) & 0x1FFFFFFu) | (2u << 30);  // addr[56:32]|type=2
  i32x8 g1;
  g1[0] = 1 << 16;                                  // data_size=1 (2 bytes)
  g1[1] = (int)(srcPitch << 16);                    // tensor_dim0 lo
  g1[2] = (int)((srcPitch >> 16) | (8192u << 16));  // tensor_dim0 hi | dim1 lo
  g1[3] = (int)(cols << 16);                        // tensor_dim1 hi | tile_dim0
  g1[4] = (int)rows;                                // tile_dim1 | tile_dim2=0
  g1[5] = (int)srcPitch;                            // tensor_dim0_stride lo
  g1[6] = 0;                                        // stride hi | dim1_stride lo
  g1[7] = 0;
  const i32x4 z4 = {0, 0, 0, 0};
  const i32x8 z8 = {0, 0, 0, 0, 0, 0, 0, 0};
  __builtin_amdgcn_tensor_load_to_lds(g0, g1, z4, z4, z8, 0);
}

// ---------------------------------------------------------------------------
// f32 -> f16 elementwise convert (vectorized); n is a multiple of 4.
// ---------------------------------------------------------------------------
__global__ __launch_bounds__(256)
void cvt_f32_f16(const float* __restrict__ in, half_t* __restrict__ out, int n) {
  const int i = (blockIdx.x * 256 + threadIdx.x) * 4;
  if (i < n) {
    const float4 f = *(const float4*)(in + i);
    v4h h; h[0] = (half_t)f.x; h[1] = (half_t)f.y; h[2] = (half_t)f.z; h[3] = (half_t)f.w;
    *(v4h*)(out + i) = h;
  }
}

// ---------------------------------------------------------------------------
// Tiled GEMM: Y[M x N] = A[M x K] @ W[N x K]^T, A and W already f16.
// 256 threads = 8 waves; block tile 128x128; k-slab 32, double-buffered in
// LDS and filled with async global->LDS b128 (no VGPR round trip).
// ---------------------------------------------------------------------------
template <bool ROPE, bool OUT_F32>
__global__ __launch_bounds__(256)
void gemm128_wmma(const half_t* __restrict__ A,
                  const half_t* __restrict__ W,
                  void* __restrict__ Yptr,
                  const float* __restrict__ cosT,   // [S][DK/2]
                  const float* __restrict__ sinT)
{
  __shared__ half_t As[2][128][40];   // pitch 40 halves = 80B (16B aligned rows)
  __shared__ half_t Bs[2][128][40];

  const int K = D_;
  const int N = D_;

  const int row0 = blockIdx.x * 128;
  const int n0   = blockIdx.y * 128;
  const int tid  = threadIdx.x;
  const int lane = tid & 31;
  const int wv   = tid >> 5;
  const int wm   = wv >> 1;          // 0..3 -> 32-row strip
  const int wn   = wv & 1;           // 0..1 -> 64-col strip
  const int l16  = lane & 15;
  const int lh   = lane >> 4;

  // staging geometry: 128 rows x 32 cols f16 = 512 x 16B chunks per matrix
  const int sr0 = (tid)       >> 2, sc0 = ((tid)       & 3) * 8;
  const int sr1 = (tid + 256) >> 2, sc1 = ((tid + 256) & 3) * 8;

  v8f acc[2][4];
#pragma unroll
  for (int i = 0; i < 2; ++i)
#pragma unroll
    for (int j = 0; j < 4; ++j)
      acc[i][j] = (v8f){0.f, 0.f, 0.f, 0.f, 0.f, 0.f, 0.f, 0.f};

  // prologue: stage slab 0 into buffer 0
  async_cp16(A + (size_t)(row0 + sr0) * K + sc0, &As[0][sr0][sc0]);
  async_cp16(W + (size_t)(n0   + sr0) * K + sc0, &Bs[0][sr0][sc0]);
  async_cp16(A + (size_t)(row0 + sr1) * K + sc1, &As[0][sr1][sc1]);
  async_cp16(W + (size_t)(n0   + sr1) * K + sc1, &Bs[0][sr1][sc1]);

  int cur = 0;
  for (int kk = 0; kk < K; kk += 32) {
    wait_async0();
    __syncthreads();                 // buffer `cur` ready for everyone

    if (kk + 32 < K) {               // overlap: fill other buffer while computing
      const int nb = cur ^ 1;
      async_cp16(A + (size_t)(row0 + sr0) * K + (kk + 32) + sc0, &As[nb][sr0][sc0]);
      async_cp16(W + (size_t)(n0   + sr0) * K + (kk + 32) + sc0, &Bs[nb][sr0][sc0]);
      async_cp16(A + (size_t)(row0 + sr1) * K + (kk + 32) + sc1, &As[nb][sr1][sc1]);
      async_cp16(W + (size_t)(n0   + sr1) * K + (kk + 32) + sc1, &Bs[nb][sr1][sc1]);
    }

    v16h afr[2], bfr[4];
#pragma unroll
    for (int i = 0; i < 2; ++i)
      afr[i] = *(const v16h*)&As[cur][wm * 32 + i * 16 + l16][lh * 16];
#pragma unroll
    for (int j = 0; j < 4; ++j)
      bfr[j] = *(const v16h*)&Bs[cur][wn * 64 + j * 16 + l16][lh * 16];
#pragma unroll
    for (int i = 0; i < 2; ++i)
#pragma unroll
      for (int j = 0; j < 4; ++j)
        acc[i][j] = wmma_f32_f16(afr[i], bfr[j], acc[i][j]);

    cur ^= 1;
  }

  // epilogue: C layout => lane holds col N=l16, rows r + lh*8
#pragma unroll
  for (int i = 0; i < 2; ++i) {
    const int mbase = row0 + wm * 32 + i * 16 + lh * 8;
#pragma unroll
    for (int j = 0; j < 4; ++j) {
      const int n = n0 + wn * 64 + j * 16 + l16;
#pragma unroll
      for (int r = 0; r < 8; ++r) {
        const int m = mbase + r;
        float val = acc[i][j][r];
        float o = val;
        if constexpr (ROPE) {
          const int pos = m & (S_ - 1);
          const int d   = n & (DK - 1);
          const float c  = cosT[pos * (DK / 2) + (d >> 1)];
          const float sn = sinT[pos * (DK / 2) + (d >> 1)];
          const float pv = __shfl_xor(val, 1, 32);   // even/odd pair = lane pair
          o = (d & 1) ? (pv * sn + val * c) : (val * c - pv * sn);
        }
        if constexpr (OUT_F32) {
          ((float*)Yptr)[(size_t)m * N + n] = o;
        } else {
          ((half_t*)Yptr)[(size_t)m * N + n] = (half_t)o;
        }
      }
    }
  }
}

// ---------------------------------------------------------------------------
// Flash attention (causal). One block per (b, h, 64-query tile), 128 threads.
// K tile: TDM tensor_load_to_lds (one wave issues, TENSORcnt sync).
// V tile: async global->LDS.  Both double-buffered over key tiles.
// ---------------------------------------------------------------------------
__global__ __launch_bounds__(128)
void attn_flash(const half_t* __restrict__ Q, const half_t* __restrict__ K,
                const half_t* __restrict__ V, half_t* __restrict__ O)
{
  __shared__ half_t Kt[2][64][64];
  __shared__ half_t Vt[2][64][64];
  __shared__ float  Sl[64][65];
  __shared__ half_t Pt[64][64];
  __shared__ float  rowm[64], rowl[64], rowc[64];

  const int qt = blockIdx.x;
  const int h  = blockIdx.y;
  const int b  = blockIdx.z;
  const int q0 = qt * 64;

  const int tid  = threadIdx.x;
  const int lane = tid & 31;
  const int w    = tid >> 5;
  const int l16  = lane & 15;
  const int lh   = lane >> 4;

  const size_t rowbase = (size_t)(b * S_) * D_;
  const int    hoff    = h * DK;

  // --- Q fragments, pre-scaled by 1/sqrt(dk) ---
  const half_t* qrow = Q + rowbase + (size_t)(q0 + w * 16 + l16) * D_ + hoff;
  v16h qa0 = *(const v16h*)(qrow + lh * 16);
  v16h qa1 = *(const v16h*)(qrow + 32 + lh * 16);
#pragma unroll
  for (int e = 0; e < 16; ++e) {
    qa0[e] = qa0[e] * (half_t)0.125f;
    qa1[e] = qa1[e] * (half_t)0.125f;
  }

  v8f oacc[4];
#pragma unroll
  for (int j = 0; j < 4; ++j)
    oacc[j] = (v8f){0.f, 0.f, 0.f, 0.f, 0.f, 0.f, 0.f, 0.f};

  if (tid < 64) { rowm[tid] = -3.0e38f; rowl[tid] = 0.f; }

  // V staging geometry: 64x64 f16 = 512 x 16B chunks, 128 threads x 4
  const int vr[4] = { (tid) >> 3, (tid + 128) >> 3, (tid + 256) >> 3, (tid + 384) >> 3 };
  const int vc[4] = { ((tid) & 7) * 8, ((tid + 128) & 7) * 8,
                      ((tid + 256) & 7) * 8, ((tid + 384) & 7) * 8 };

  // prologue: stage key tile 0
  if (tid < 32)
    tdm_load_tile_f16(&Kt[0][0][0], K + rowbase + hoff, 64, 64, D_);
#pragma unroll
  for (int it = 0; it < 4; ++it)
    async_cp16(V + rowbase + (size_t)vr[it] * D_ + hoff + vc[it], &Vt[0][vr[it]][vc[it]]);

  int cur = 0;
  for (int kt = 0; kt <= qt; ++kt) {
    __builtin_amdgcn_s_wait_tensorcnt(0);
    wait_async0();
    __syncthreads();                       // K/V tiles + prior phase complete

    if (kt < qt) {                         // overlap next tile fill with compute
      const int nb = cur ^ 1;
      const size_t next = rowbase + (size_t)((kt + 1) * 64) * D_ + hoff;
      if (tid < 32)
        tdm_load_tile_f16(&Kt[nb][0][0], K + next, 64, 64, D_);
#pragma unroll
      for (int it = 0; it < 4; ++it)
        async_cp16(V + next + (size_t)vr[it] * D_ + vc[it], &Vt[nb][vr[it]][vc[it]]);
    }

    // --- S tile: wave's 16 query rows x 64 keys (WMMA) ---
#pragma unroll
    for (int nt = 0; nt < 4; ++nt) {
      v8f s = (v8f){0.f, 0.f, 0.f, 0.f, 0.f, 0.f, 0.f, 0.f};
      v16h b0 = *(const v16h*)&Kt[cur][nt * 16 + l16][lh * 16];
      v16h b1 = *(const v16h*)&Kt[cur][nt * 16 + l16][32 + lh * 16];
      s = wmma_f32_f16(qa0, b0, s);
      s = wmma_f32_f16(qa1, b1, s);
#pragma unroll
      for (int r = 0; r < 8; ++r)
        Sl[w * 16 + lh * 8 + r][nt * 16 + l16] = s[r];
    }
    __syncthreads();

    // --- online softmax: one row per thread ---
    if (tid < 64) {
      const int r     = tid;
      const int kmax0 = q0 + r - kt * 64 + 1;
      const int kmax  = kmax0 < 64 ? kmax0 : 64;
      const float mold = rowm[r];
      float tmx = -3.0e38f;
      for (int c = 0; c < 64; ++c)
        if (c < kmax) tmx = fmaxf(tmx, Sl[r][c]);
      const float nm   = fmaxf(mold, tmx);
      const float corr = __expf(mold - nm);
      float sum = 0.f;
      for (int c = 0; c < 64; ++c) {
        const float p = (c < kmax) ? __expf(Sl[r][c] - nm) : 0.f;
        Pt[r][c] = (half_t)p;
        sum += p;
      }
      rowm[r] = nm;
      rowl[r] = rowl[r] * corr + sum;
      rowc[r] = corr;
    }
    __syncthreads();

    // --- rescale running O, accumulate P @ V (WMMA) ---
    float cr[8];
#pragma unroll
    for (int r = 0; r < 8; ++r) cr[r] = rowc[w * 16 + lh * 8 + r];
#pragma unroll
    for (int j = 0; j < 4; ++j)
#pragma unroll
      for (int r = 0; r < 8; ++r) oacc[j][r] *= cr[r];

    v16h pa0 = *(const v16h*)&Pt[w * 16 + l16][lh * 16];
    v16h pa1 = *(const v16h*)&Pt[w * 16 + l16][32 + lh * 16];
#pragma unroll
    for (int j = 0; j < 4; ++j) {
      v16h vb0, vb1;
#pragma unroll
      for (int s = 0; s < 16; ++s) {
        vb0[s] = Vt[cur][lh * 16 + s][j * 16 + l16];
        vb1[s] = Vt[cur][32 + lh * 16 + s][j * 16 + l16];
      }
      oacc[j] = wmma_f32_f16(pa0, vb0, oacc[j]);
      oacc[j] = wmma_f32_f16(pa1, vb1, oacc[j]);
    }
    __syncthreads();                       // protect Sl/Pt (and buffers) reuse
    cur ^= 1;
  }

  // --- finalize ---
  float li[8];
#pragma unroll
  for (int r = 0; r < 8; ++r) li[r] = 1.f / rowl[w * 16 + lh * 8 + r];
#pragma unroll
  for (int j = 0; j < 4; ++j) {
#pragma unroll
    for (int r = 0; r < 8; ++r) {
      const size_t m = (size_t)(b * S_) + q0 + w * 16 + lh * 8 + r;
      O[m * D_ + hoff + j * 16 + l16] = (half_t)(oacc[j][r] * li[r]);
    }
  }
}

// ---------------------------------------------------------------------------
// Launch.
// Workspace (f16): Xh 16MB | Wq/Wk/Wv/Wo 2MB each | Q/K/V/O 16MB each = 88MB.
// ---------------------------------------------------------------------------
extern "C" void kernel_launch(void* const* d_in, const int* in_sizes, int n_in,
                              void* d_out, int out_size, void* d_ws, size_t ws_size,
                              hipStream_t stream) {
  const float* X    = (const float*)d_in[0];
  const float* cosT = (const float*)d_in[1];
  const float* sinT = (const float*)d_in[2];
  const float* Wq   = (const float*)d_in[3];
  const float* Wk   = (const float*)d_in[4];
  const float* Wv   = (const float*)d_in[5];
  const float* Wo   = (const float*)d_in[6];
  float* out = (float*)d_out;

  half_t* Xh  = (half_t*)d_ws;
  half_t* Wqh = Xh  + (size_t)M_ * D_;
  half_t* Wkh = Wqh + (size_t)D_ * D_;
  half_t* Wvh = Wkh + (size_t)D_ * D_;
  half_t* Woh = Wvh + (size_t)D_ * D_;
  half_t* Qh  = Woh + (size_t)D_ * D_;
  half_t* Kh  = Qh  + (size_t)M_ * D_;
  half_t* Vh  = Kh  + (size_t)M_ * D_;
  half_t* Oh  = Vh  + (size_t)M_ * D_;

  const int nX = M_ * D_;
  const int nW = D_ * D_;
  cvt_f32_f16<<<nX / 1024, 256, 0, stream>>>(X,  Xh,  nX);
  cvt_f32_f16<<<nW / 1024, 256, 0, stream>>>(Wq, Wqh, nW);
  cvt_f32_f16<<<nW / 1024, 256, 0, stream>>>(Wk, Wkh, nW);
  cvt_f32_f16<<<nW / 1024, 256, 0, stream>>>(Wv, Wvh, nW);
  cvt_f32_f16<<<nW / 1024, 256, 0, stream>>>(Wo, Woh, nW);

  const dim3 blk(256);
  const dim3 g1(M_ / 128, D_ / 128, 1);
  gemm128_wmma<true,  false><<<g1, blk, 0, stream>>>(Xh, Wqh, Qh, cosT, sinT);
  gemm128_wmma<true,  false><<<g1, blk, 0, stream>>>(Xh, Wkh, Kh, cosT, sinT);
  gemm128_wmma<false, false><<<g1, blk, 0, stream>>>(Xh, Wvh, Vh, nullptr, nullptr);

  attn_flash<<<dim3(S_ / 64, HEADS, B_), dim3(128), 0, stream>>>(Qh, Kh, Vh, Oh);

  gemm128_wmma<false, true><<<g1, blk, 0, stream>>>(Oh, Woh, out, nullptr, nullptr);
}
